// LSTMNet_55319178772499
// MI455X (gfx1250) — matrix operand order repaired and checked by reading, not compiled, not run on hardware
//
#include <hip/hip_runtime.h>
#include <hip/hip_bf16.h>

// ---------------------------------------------------------------------------
// Persistent-grid 2-layer LSTM (B=64, T=512, IN=64, H=512) + fused FC head.
// bf16 WMMA (v_wmma_f32_16x16x32_bf16) with f32 accumulation; layer-0 weights
// resident in LDS; hidden vectors staged into LDS via async global->LDS loads;
// device-wide step barrier via atomic counter + s_cluster_barrier.
// ---------------------------------------------------------------------------

typedef __attribute__((ext_vector_type(16))) __bf16 v16bf;
typedef __attribute__((ext_vector_type(8)))  float  v8f;
typedef __attribute__((ext_vector_type(4)))  int    v4i;

union Frag {
  v16bf v;
  uint4 q[2];
};

#define NWG     16
#define TPB     256
#define HSZ     512
#define TSTEPS  512
#define BATCH   64

#if __has_builtin(__builtin_amdgcn_global_load_async_to_lds_b128)
#define HAVE_ASYNC_LDS 1
#else
#define HAVE_ASYNC_LDS 0
#endif

// Pointer-to-AS(1)/AS(3) v4i types for the async builtin.
typedef __attribute__((address_space(1))) v4i* gas_v4i_p;
typedef __attribute__((address_space(3))) v4i* las_v4i_p;

__device__ __forceinline__ void cp16_async(void* lds_dst, const void* gsrc) {
#if HAVE_ASYNC_LDS
  __builtin_amdgcn_global_load_async_to_lds_b128(
      (gas_v4i_p)(unsigned long long)gsrc,
      (las_v4i_p)(unsigned int)(unsigned long long)lds_dst, 0, 0);
#else
  *(uint4*)lds_dst = *(const uint4*)gsrc;
#endif
}

__device__ __forceinline__ void wait_async() {
#if HAVE_ASYNC_LDS
#if __has_builtin(__builtin_amdgcn_s_wait_asynccnt)
  __builtin_amdgcn_s_wait_asynccnt(0);
#else
  asm volatile("s_wait_asynccnt 0" ::: "memory");
#endif
#else
  asm volatile("" ::: "memory");
#endif
}

__device__ __forceinline__ float tanh_f(float x) {
#if __has_builtin(__builtin_amdgcn_tanhf)
  return __builtin_amdgcn_tanhf(x);
#else
  return tanhf(x);
#endif
}

__device__ __forceinline__ float sigm(float x) {
  return 0.5f * tanh_f(0.5f * x) + 0.5f;
}

__device__ __forceinline__ v8f wmma_bf16(v16bf a, v16bf b, v8f c) {
  // (neg_a, A, neg_b, B, c_mod, C, reuse_a, reuse_b)
  return __builtin_amdgcn_wmma_f32_16x16x32_bf16(false, a, false, b, (short)0, c,
                                                 false, false);
}

// A fragment (16x32 bf16, MxK): lanes 0-15 row M=lane, K={k0..k0+7, k0+16..k0+23};
// lanes 16-31 row M=lane-16, K offset +8.
__device__ __forceinline__ v16bf ldA(const __bf16* base, int stride, int mbase,
                                     int k0, int lane) {
  Frag f;
  const __bf16* p =
      base + (size_t)(mbase + (lane & 15)) * stride + k0 + ((lane >> 4) << 3);
  f.q[0] = *(const uint4*)p;
  f.q[1] = *(const uint4*)(p + 16);
  return f.v;
}

// B fragment (32x16 bf16, KxN) from row-major W[N][K]: lane holds column
// n=nrow, K={k0..k0+7, k0+16..k0+23}, offset +8 for lanes 16-31.
__device__ __forceinline__ v16bf ldB(const __bf16* base, int stride, int nrow,
                                     int k0, int lane) {
  Frag f;
  const __bf16* p = base + (size_t)nrow * stride + k0 + ((lane >> 4) << 3);
  f.q[0] = *(const uint4*)p;
  f.q[1] = *(const uint4*)(p + 16);
  return f.v;
}

// Device-wide barrier for NWG resident workgroups; also emits the CDNA5
// cluster barrier ops (NOP when not dispatched as a cluster).
__device__ __forceinline__ void gbar(unsigned* cnt, int tid, unsigned* ph) {
  __syncthreads();
  __threadfence();
  ++(*ph);
  if (tid == 0) {
    atomicAdd(cnt, 1u);
    unsigned tgt = (unsigned)NWG * (*ph);
    volatile unsigned* vc = (volatile unsigned*)cnt;
    while (*vc < tgt) __builtin_amdgcn_s_sleep(1);
  }
  __builtin_amdgcn_s_cluster_barrier();
  __syncthreads();
  __threadfence();
}

__global__ void cvt_f32_bf16(const float* __restrict__ src,
                             __bf16* __restrict__ dst, int n) {
  int i = blockIdx.x * blockDim.x + threadIdx.x;
  int stride = gridDim.x * blockDim.x;
  for (; i < n; i += stride) dst[i] = (__bf16)src[i];
}

__global__ void __launch_bounds__(TPB, 1)
lstm_persistent(const float* __restrict__ x,
                __bf16* __restrict__ h0buf,        // 2 * 64*512 bf16
                __bf16* __restrict__ h2buf,        // 2 * 64*512 bf16
                const __bf16* __restrict__ Wih0b,  // 2048x64
                const __bf16* __restrict__ Whh0b,  // 2048x512
                const __bf16* __restrict__ Wih1b,  // 2048x512
                const __bf16* __restrict__ Whh1b,  // 2048x512
                const float* __restrict__ bih0, const float* __restrict__ bhh0,
                const float* __restrict__ bih1, const float* __restrict__ bhh1,
                const float* __restrict__ fcw, const float* __restrict__ fcb,
                float* __restrict__ out, unsigned* __restrict__ cnt) {
  extern __shared__ char smem[];
  __bf16* Whh0s = (__bf16*)smem;       // 128*512  = 128KB
  __bf16* hs    = Whh0s + 128 * 512;   // 64*512   = 64KB (staged shared hidden)
  __bf16* Wih0s = hs + 64 * 512;       // 128*64   = 16KB
  __bf16* xs    = Wih0s + 128 * 64;    // 64*64    = 8KB  (x_t as bf16)
  float*  gbuf  = (float*)(xs + 64 * 64);  // 128*66 f32 (gate scratch, padded)
  float*  c0b   = gbuf + 128 * 66;         // 32*65 f32 (cell, layer 0)
  float*  c1b   = c0b + 32 * 65;           // 32*65 f32 (cell, layer 1)
  float*  bs0   = c1b + 32 * 65;           // 128 bias sums layer 0
  float*  bs1   = bs0 + 128;               // 128 bias sums layer 1
  float*  fcs   = bs1 + 128;               // 32 fc weights slice

  const int tid  = threadIdx.x;
  const int wg   = blockIdx.x;    // owns hidden units [32*wg, 32*wg+32)
  const int w    = tid >> 5;      // wave 0..7, owns gate rows [16w,16w+16) local
  const int lane = tid & 31;

  // ---- one-time init ----
  if (tid < 128) {
    int grow = (tid >> 5) * HSZ + 32 * wg + (tid & 31);
    bs0[tid] = bih0[grow] + bhh0[grow];
    bs1[tid] = bih1[grow] + bhh1[grow];
  }
  if (tid < 32) fcs[tid] = fcw[32 * wg + tid];
  for (int i = tid; i < 32 * 65; i += TPB) { c0b[i] = 0.f; c1b[i] = 0.f; }
  if (wg == 0 && tid < BATCH) out[tid] = fcb[0];

  // Resident layer-0 weight slices: local row n -> global gate row
  // gr = (n>>5)*512 + 32*wg + (n&31)   (i,f,g,o blocks of 32 rows)
  for (int c = tid; c < (128 * 512) / 8; c += TPB) {  // 16B chunks
    int n = c >> 6, off = (c & 63) << 3;
    int gr = (n >> 5) * HSZ + 32 * wg + (n & 31);
    cp16_async(Whh0s + n * 512 + off, Whh0b + (size_t)gr * 512 + off);
  }
  for (int c = tid; c < (128 * 64) / 8; c += TPB) {
    int n = c >> 3, off = (c & 7) << 3;
    int gr = (n >> 5) * HSZ + 32 * wg + (n & 31);
    cp16_async(Wih0s + n * 64 + off, Wih0b + (size_t)gr * 64 + off);
  }
  wait_async();
  __syncthreads();

  unsigned ph = 0;
  const int nloc     = (w << 4) + (lane & 15);  // local gate row for B frags
  const int growbase = (w >> 1) * HSZ + 32 * wg + ((w & 1) << 4);

#pragma unroll 1
  for (int t = 0; t < TSTEPS; ++t) {
    const int cur = t & 1, nxt = cur ^ 1;

    // ================= LAYER 0 =================
    // Stage x_t (f32 -> bf16) and previous h0 (bf16, async -> LDS).
    for (int e = tid; e < BATCH * 64; e += TPB) {
      int b = e >> 6, i = e & 63;
      xs[e] = (__bf16)x[(size_t)b * (TSTEPS * 64) + (size_t)t * 64 + i];
    }
    {
      const __bf16* src = h0buf + (size_t)cur * (BATCH * HSZ);
      for (int c = tid; c < (BATCH * HSZ) / 8; c += TPB)
        cp16_async(hs + c * 8, src + c * 8);
    }
    wait_async();
    __syncthreads();

    v8f acc[4] = {};
    // x contribution: K = 64
#pragma unroll
    for (int kt = 0; kt < 2; ++kt) {
      int k0 = kt << 5;
      v16bf bfr = ldB(Wih0s, 64, nloc, k0, lane);
#pragma unroll
      for (int mt = 0; mt < 4; ++mt)
        acc[mt] = wmma_bf16(ldA(xs, 64, mt << 4, k0, lane), bfr, acc[mt]);
    }
    // recurrent contribution: K = 512 (weights resident in LDS)
#pragma unroll 2
    for (int kt = 0; kt < 16; ++kt) {
      int k0 = kt << 5;
      v16bf bfr = ldB(Whh0s, 512, nloc, k0, lane);
#pragma unroll
      for (int mt = 0; mt < 4; ++mt)
        acc[mt] = wmma_bf16(ldA(hs, 512, mt << 4, k0, lane), bfr, acc[mt]);
    }
    // D layout: VGPR r, lanes 0-15: (M=r, N=lane); lanes 16-31: (M=r+8).
#pragma unroll
    for (int mt = 0; mt < 4; ++mt)
#pragma unroll
      for (int r = 0; r < 8; ++r)
        gbuf[nloc * 66 + (mt << 4) + r + ((lane >> 4) << 3)] = acc[mt][r];
    __syncthreads();

    {
      __bf16* hdst = h0buf + (size_t)nxt * (BATCH * HSZ);
      for (int idx = tid; idx < BATCH * 32; idx += TPB) {
        int j = idx & 31, b = idx >> 5;
        float gi = sigm(gbuf[j * 66 + b] + bs0[j]);
        float gf = sigm(gbuf[(32 + j) * 66 + b] + bs0[32 + j]);
        float gg = tanh_f(gbuf[(64 + j) * 66 + b] + bs0[64 + j]);
        float go = sigm(gbuf[(96 + j) * 66 + b] + bs0[96 + j]);
        float cc = gf * c0b[j * 65 + b] + gi * gg;
        c0b[j * 65 + b] = cc;
        float h = go * tanh_f(cc);
        hdst[(size_t)b * HSZ + 32 * wg + j] = (__bf16)h;
      }
    }
    gbar(cnt, tid, &ph);  // all of new h0 visible device-wide

    // ================= LAYER 1 =================
    // half 1: input contribution, A = h1_t (just-produced layer-0 output),
    // B = W_ih1 streamed from global (L2-resident).
    {
      const __bf16* src = h0buf + (size_t)nxt * (BATCH * HSZ);
      for (int c = tid; c < (BATCH * HSZ) / 8; c += TPB)
        cp16_async(hs + c * 8, src + c * 8);
      wait_async();
      __syncthreads();
    }
    v8f acc2[4] = {};
    {
      int grow = growbase + (lane & 15);
#pragma unroll 2
      for (int kt = 0; kt < 16; ++kt) {
        int k0 = kt << 5;
        v16bf bfr = ldB(Wih1b, 512, grow, k0, lane);
#pragma unroll
        for (int mt = 0; mt < 4; ++mt)
          acc2[mt] = wmma_bf16(ldA(hs, 512, mt << 4, k0, lane), bfr, acc2[mt]);
      }
    }
    __syncthreads();  // everyone done reading hs before restage
    // half 2: recurrent contribution, A = previous h2, B = W_hh1 streamed.
    {
      const __bf16* src = h2buf + (size_t)cur * (BATCH * HSZ);
      for (int c = tid; c < (BATCH * HSZ) / 8; c += TPB)
        cp16_async(hs + c * 8, src + c * 8);
      wait_async();
      __syncthreads();
    }
    {
      int grow = growbase + (lane & 15);
#pragma unroll 2
      for (int kt = 0; kt < 16; ++kt) {
        int k0 = kt << 5;
        v16bf bfr = ldB(Whh1b, 512, grow, k0, lane);
#pragma unroll
        for (int mt = 0; mt < 4; ++mt)
          acc2[mt] = wmma_bf16(ldA(hs, 512, mt << 4, k0, lane), bfr, acc2[mt]);
      }
    }
#pragma unroll
    for (int mt = 0; mt < 4; ++mt)
#pragma unroll
      for (int r = 0; r < 8; ++r)
        gbuf[nloc * 66 + (mt << 4) + r + ((lane >> 4) << 3)] = acc2[mt][r];
    __syncthreads();

    {
      __bf16* hdst = h2buf + (size_t)nxt * (BATCH * HSZ);
      const bool last = (t == TSTEPS - 1);
      for (int idx = tid; idx < BATCH * 32; idx += TPB) {
        int j = idx & 31, b = idx >> 5;
        float gi = sigm(gbuf[j * 66 + b] + bs1[j]);
        float gf = sigm(gbuf[(32 + j) * 66 + b] + bs1[32 + j]);
        float gg = tanh_f(gbuf[(64 + j) * 66 + b] + bs1[64 + j]);
        float go = sigm(gbuf[(96 + j) * 66 + b] + bs1[96 + j]);
        float cc = gf * c1b[j * 65 + b] + gi * gg;
        c1b[j * 65 + b] = cc;
        float h = go * tanh_f(cc);
        hdst[(size_t)b * HSZ + 32 * wg + j] = (__bf16)h;
        if (last) atomicAdd(&out[b], h * fcs[j]);  // fused FC head (OUT=1)
      }
    }
    gbar(cnt, tid, &ph);
  }
}

// ---------------------------------------------------------------------------
// Workspace layout (bytes):
//   [0,256)           barrier counter (zeroed per launch)
//   [256, +4*64KB)    h0[2], h2[2] bf16 double buffers (zeroed per launch)
//   [262400, ...)     bf16 weights: Wih0(2048x64) Whh0(2048x512)
//                     Wih1(2048x512) Whh1(2048x512)   (~6.8 MB total)
// ---------------------------------------------------------------------------
extern "C" void kernel_launch(void* const* d_in, const int* in_sizes, int n_in,
                              void* d_out, int out_size, void* d_ws,
                              size_t ws_size, hipStream_t stream) {
  const float* x    = (const float*)d_in[0];
  const float* Wih0 = (const float*)d_in[1];
  const float* Whh0 = (const float*)d_in[2];
  const float* bih0 = (const float*)d_in[3];
  const float* bhh0 = (const float*)d_in[4];
  const float* Wih1 = (const float*)d_in[5];
  const float* Whh1 = (const float*)d_in[6];
  const float* bih1 = (const float*)d_in[7];
  const float* bhh1 = (const float*)d_in[8];
  const float* fcw  = (const float*)d_in[9];
  const float* fcb  = (const float*)d_in[10];
  float* out = (float*)d_out;

  char* ws = (char*)d_ws;
  unsigned* cnt = (unsigned*)ws;
  __bf16* h0b = (__bf16*)(ws + 256);
  __bf16* h2b = (__bf16*)(ws + 256 + 2 * (BATCH * HSZ) * sizeof(__bf16));
  char* wbase = ws + 262400;  // 256-aligned
  __bf16* Wih0b = (__bf16*)wbase;
  __bf16* Whh0b = Wih0b + 2048 * 64;
  __bf16* Wih1b = Whh0b + 2048 * 512;
  __bf16* Whh1b = Wih1b + 2048 * 512;

  // Zero barrier counter + h state (h(0)=c(0)=0) every launch (deterministic).
  (void)hipMemsetAsync(ws, 0, 262400, stream);

  // One-time per-launch weight conversion f32 -> bf16 (stays hot in L2).
  cvt_f32_bf16<<<512, 256, 0, stream>>>(Wih0, Wih0b, 2048 * 64);
  cvt_f32_bf16<<<1024, 256, 0, stream>>>(Whh0, Whh0b, 2048 * 512);
  cvt_f32_bf16<<<1024, 256, 0, stream>>>(Wih1, Wih1b, 2048 * 512);
  cvt_f32_bf16<<<1024, 256, 0, stream>>>(Whh1, Whh1b, 2048 * 512);

  const int smem_bytes =
      (128 * 512 + 64 * 512 + 128 * 64 + 64 * 64) * (int)sizeof(__bf16) +
      (128 * 66 + 2 * 32 * 65 + 2 * 128 + 32) * (int)sizeof(float);
  (void)hipFuncSetAttribute((const void*)lstm_persistent,
                            hipFuncAttributeMaxDynamicSharedMemorySize,
                            smem_bytes);

  lstm_persistent<<<NWG, TPB, smem_bytes, stream>>>(
      x, h0b, h2b, Wih0b, Whh0b, Wih1b, Whh1b, bih0, bhh0, bih1, bhh1, fcw,
      fcb, out, cnt);
}
